// StateSpaceLayer_47717086659301
// MI455X (gfx1250) — matrix operand order repaired and checked by reading, not compiled, not run on hardware
//
#include <hip/hip_runtime.h>

typedef __attribute__((ext_vector_type(16))) _Float16 v16h;
typedef __attribute__((ext_vector_type(2)))  _Float16 v2h_t;
typedef __attribute__((ext_vector_type(2)))  __fp16   v2fp16;
typedef __attribute__((ext_vector_type(8)))  float    v8f;
typedef int v4i_ __attribute__((vector_size(16)));   // matches builtin param pointee

#define SEQ 1024
#define DIM 64
#define NELEM (DIM * DIM)   // 4096 per timestep

#if defined(__has_builtin)
#if __has_builtin(__builtin_amdgcn_global_load_async_to_lds_b128) && \
    __has_builtin(__builtin_amdgcn_s_wait_asynccnt)
#define SSM_ASYNC_LDS 1
#endif
#endif

// Pack two f32 -> packed f16 pair via v_cvt_pk_rtz_f16_f32.
static __device__ __forceinline__ v2h_t pk2(float a, float b) {
    v2fp16 t = __builtin_amdgcn_cvt_pkrtz(a, b);
    return __builtin_bit_cast(v2h_t, t);
}

// ---------------------------------------------------------------------------
// Kernel 1: LayerNorm over (d,d) + dt projection + softplus + log_a
// One block per timestep t. Outputs row-major f16 x_norm and log_a[t][r].
// ---------------------------------------------------------------------------
__global__ __launch_bounds__(256) void ln_dt_kernel(
    const float* __restrict__ x,
    const float* __restrict__ log_A,
    const float* __restrict__ dt_w,
    const float* __restrict__ dt_b,
    const float* __restrict__ ln_w,
    const float* __restrict__ ln_b,
    _Float16*    __restrict__ Xh,       // [t][r*64+c] f16
    float*       __restrict__ log_a)    // [t][r]
{
    __shared__ float xc[NELEM];     // normalized values (16 KB)
    __shared__ float red[256];
    __shared__ float stats[2];      // mean, rstd

    const int t   = blockIdx.x;
    const int tid = threadIdx.x;
    const float* xb = x + (size_t)t * NELEM;

    float vals[16];
    float s = 0.f, s2 = 0.f;
#pragma unroll
    for (int i = 0; i < 4; ++i) {
        float4 v = ((const float4*)xb)[tid * 4 + i];
        vals[i*4+0] = v.x; vals[i*4+1] = v.y; vals[i*4+2] = v.z; vals[i*4+3] = v.w;
        s  += v.x + v.y + v.z + v.w;
        s2 += v.x*v.x + v.y*v.y + v.z*v.z + v.w*v.w;
    }

    red[tid] = s; __syncthreads();
    for (int o = 128; o > 0; o >>= 1) {
        if (tid < o) red[tid] += red[tid + o];
        __syncthreads();
    }
    if (tid == 0) stats[0] = red[0] * (1.0f / NELEM);
    __syncthreads();

    red[tid] = s2; __syncthreads();
    for (int o = 128; o > 0; o >>= 1) {
        if (tid < o) red[tid] += red[tid + o];
        __syncthreads();
    }
    if (tid == 0) {
        float mean = stats[0];
        float var  = red[0] * (1.0f / NELEM) - mean * mean;  // biased variance
        stats[1]   = rsqrtf(var + 1e-5f);
    }
    __syncthreads();

    const float mean = stats[0];
    const float rstd = stats[1];

#pragma unroll
    for (int i = 0; i < 16; ++i) {
        int   idx = tid * 16 + i;
        float xn  = (vals[i] - mean) * rstd * ln_w[idx] + ln_b[idx];
        xc[idx] = xn;
        Xh[(size_t)t * NELEM + idx] = (_Float16)xn;
    }
    __syncthreads();

    if (tid < DIM) {
        const int r = tid;
        float dot = 0.f;
#pragma unroll 8
        for (int c = 0; c < DIM; ++c) dot += xc[r * DIM + c] * dt_w[c];
        dot += dt_b[0];
        float dt = (dot > 20.f) ? dot : log1pf(__expf(dot));  // softplus
        log_a[(size_t)t * DIM + r] = -__expf(log_A[r]) * dt;
    }
}

// ---------------------------------------------------------------------------
// Kernel 2: per-channel inclusive cumsum over seq (Hillis-Steele in LDS).
// Output: cums channel-major [r][t]
// ---------------------------------------------------------------------------
__global__ __launch_bounds__(1024) void scan_kernel(
    const float* __restrict__ log_a,
    float*       __restrict__ cums)
{
    __shared__ float buf[SEQ];
    const int r = blockIdx.x;
    const int t = threadIdx.x;

    float v = log_a[(size_t)t * DIM + r];
    buf[t] = v;
    __syncthreads();
#pragma unroll
    for (int off = 1; off < SEQ; off <<= 1) {
        float add = (t >= off) ? buf[t - off] : 0.0f;
        __syncthreads();
        v += add;
        buf[t] = v;
        __syncthreads();
    }
    cums[(size_t)r * SEQ + t] = v;
}

// ---------------------------------------------------------------------------
// Kernel 2b: transpose Xh [t][rc] -> XhT [rc][t]  (f16, LDS tiled).
// Grid (16, 64): 64x64 tiles. Makes B operands contiguous in memory.
// ---------------------------------------------------------------------------
__global__ __launch_bounds__(256) void transpose_kernel(
    const _Float16* __restrict__ Xh,
    _Float16*       __restrict__ XhT)
{
    __shared__ __align__(16) _Float16 tile[64][72];  // 144B row stride (16B mult)
    const int tb  = blockIdx.x;   // t-tile   (0..15)
    const int cb  = blockIdx.y;   // rc-tile  (0..63)
    const int tid = threadIdx.x;

    {   // coalesced read of 64 rows x 64 cols, scattered LDS store (transpose)
        const int tt = tid >> 2;           // 0..63
        const int c0 = (tid & 3) * 16;     // 0,16,32,48
        const _Float16* src = Xh + (size_t)(tb * 64 + tt) * NELEM + cb * 64 + c0;
        _Float16 h[16];
        *((uint4*)h)     = ((const uint4*)src)[0];
        *((uint4*)h + 1) = ((const uint4*)src)[1];
#pragma unroll
        for (int i = 0; i < 16; ++i) tile[c0 + i][tt] = h[i];
    }
    __syncthreads();
    {   // contiguous LDS read, coalesced global store
        const int cc = tid >> 2;           // 0..63
        const int t0 = (tid & 3) * 16;     // 0,16,32,48
        const uint4* lsrc = (const uint4*)&tile[cc][t0];
        uint4 u0 = lsrc[0], u1 = lsrc[1];
        _Float16* dst = XhT + (size_t)(cb * 64 + cc) * SEQ + tb * 64 + t0;
        ((uint4*)dst)[0] = u0;
        ((uint4*)dst)[1] = u1;
    }
}

// ---------------------------------------------------------------------------
// Kernel 3: out[t,r,:] = sum_{j<=t} exp(cums[t,r]-cums[j,r]) * xn[j,r,:]
// v_wmma_f32_16x16x32_f16; 1 wave/block, 2 t-tiles (32 rows) x 64 cols.
// A (decay tile) generated in registers with v_exp_f32 (factored) +
// v_cvt_pk_rtz_f16_f32; B loaded straight from transposed global memory
// as 2x global_load_b128 per tile. cums column preloaded to LDS (async
// global->LDS path when available).
// ---------------------------------------------------------------------------
__global__ __launch_bounds__(32) void ssm_wmma_kernel(
    const _Float16* __restrict__ XhT,   // [(r*64+c)][t]
    const float*    __restrict__ cums,  // [r][t]
    float*          __restrict__ out)
{
    __shared__ __align__(16) float cc[SEQ];   // 4 KB cums column

    const int lane = threadIdx.x;
    const int T0   = blockIdx.x * 32;         // 32 output rows per wave
    const int r    = blockIdx.y;

#ifdef SSM_ASYNC_LDS
    {   // each lane copies 128B: 8x async global->LDS b128, same imm offset on both sides
        typedef __attribute__((address_space(1))) v4i_ gas_v4i;
        typedef __attribute__((address_space(3))) v4i_ las_v4i;
        gas_v4i* g = (gas_v4i*)(unsigned long long)(cums + (size_t)r * SEQ + lane * 32);
        las_v4i* l = (las_v4i*)(unsigned int)(unsigned long long)(&cc[lane * 32]);
        __builtin_amdgcn_global_load_async_to_lds_b128(g, l,   0, 0);
        __builtin_amdgcn_global_load_async_to_lds_b128(g, l,  16, 0);
        __builtin_amdgcn_global_load_async_to_lds_b128(g, l,  32, 0);
        __builtin_amdgcn_global_load_async_to_lds_b128(g, l,  48, 0);
        __builtin_amdgcn_global_load_async_to_lds_b128(g, l,  64, 0);
        __builtin_amdgcn_global_load_async_to_lds_b128(g, l,  80, 0);
        __builtin_amdgcn_global_load_async_to_lds_b128(g, l,  96, 0);
        __builtin_amdgcn_global_load_async_to_lds_b128(g, l, 112, 0);
        __builtin_amdgcn_s_wait_asynccnt(0);
    }
#else
#pragma unroll 4
    for (int i = 0; i < SEQ / 32; ++i)
        cc[i * 32 + lane] = cums[(size_t)r * SEQ + i * 32 + lane];
#endif
    __syncthreads();

    const bool hi   = lane >= 16;
    const int  tl   = lane & 15;
    const int  koff = hi ? 16 : 0;   // A layout: lanes 0-15 K0..15, lanes 16-31 K16..31
    const int  kb   = hi ? 16 : 0;   // B layout: lanes 0-15 K0..15, lanes 16-31 K16..31
    const int  tg0  = T0 + tl;
    const int  tg1  = T0 + 16 + tl;
    const float ct0 = cc[tg0];
    const float ct1 = cc[tg1];

    v8f acc[8] = {};   // [ttile*4 + ctile]

    const _Float16* bbase = XhT + (size_t)(r * 64 + tl) * SEQ;

    auto kstep = [&](int J0, bool masked) {
        // per-lane K columns: cj[s] = cums[J0 + k(s)], two contiguous runs of 8
        const float cbase = cc[J0];
        float ef[16];
#pragma unroll
        for (int s = 0; s < 8; ++s) {
            ef[s]     = __expf(cbase - cc[J0 + koff + s]);
            ef[8 + s] = __expf(cbase - cc[J0 + 8 + koff + s]);
        }
        const float f0 = __expf(ct0 - cbase);   // <= 1, underflow->0 is exact for f16 A
        const float f1 = __expf(ct1 - cbase);

        v16h a0, a1;
#pragma unroll
        for (int p = 0; p < 8; ++p) {
            const int s0 = 2 * p, s1 = 2 * p + 1;
            float x00 = f0 * ef[s0], x01 = f0 * ef[s1];
            float x10 = f1 * ef[s0], x11 = f1 * ef[s1];
            if (masked) {
                const int k0 = s0 + koff;
                const int k1 = s1 + koff;
                const int j0 = J0 + k0, j1 = J0 + k1;
                x00 = (j0 <= tg0) ? x00 : 0.f;
                x01 = (j1 <= tg0) ? x01 : 0.f;
                x10 = (j0 <= tg1) ? x10 : 0.f;
                x11 = (j1 <= tg1) ? x11 : 0.f;
            }
            v2h_t q0 = pk2(x00, x01);
            v2h_t q1 = pk2(x10, x11);
            a0[s0] = q0[0]; a0[s1] = q0[1];
            a1[s0] = q1[0]; a1[s1] = q1[1];
        }

#pragma unroll
        for (int ct = 0; ct < 4; ++ct) {
            const _Float16* bp = bbase + (size_t)(ct * 16) * SEQ + J0 + kb;
            v16h b;
            *((uint4*)&b)     = ((const uint4*)bp)[0];   // K kb..kb+7
            *((uint4*)&b + 1) = ((const uint4*)bp)[1];   // K kb+8..kb+15
            acc[ct]     = __builtin_amdgcn_wmma_f32_16x16x32_f16(
                              false, a0, false, b, (short)0, acc[ct],     false, false);
            acc[4 + ct] = __builtin_amdgcn_wmma_f32_16x16x32_f16(
                              false, a1, false, b, (short)0, acc[4 + ct], false, false);
        }
    };

    const int nJ = T0 / 32 + 1;          // K-tiles covering j <= T0+31
    for (int jt = 0; jt < nJ - 1; ++jt)  // interior tiles: j <= T0-1 < t always
        kstep(jt * 32, false);
    kstep((nJ - 1) * 32, true);          // diagonal tile: causal mask

    // Store D tiles: lanes 0-15: N=lane, VGPR v -> M=v; lanes 16-31: N=lane-16, M=v+8.
#pragma unroll
    for (int tt = 0; tt < 2; ++tt) {
#pragma unroll
        for (int v = 0; v < 8; ++v) {
            const int t = T0 + tt * 16 + v + (hi ? 8 : 0);
            float* o = out + ((size_t)t * DIM + r) * DIM + tl;
            o[ 0] = acc[tt * 4 + 0][v];
            o[16] = acc[tt * 4 + 1][v];
            o[32] = acc[tt * 4 + 2][v];
            o[48] = acc[tt * 4 + 3][v];
        }
    }
}

// ---------------------------------------------------------------------------
extern "C" void kernel_launch(void* const* d_in, const int* in_sizes, int n_in,
                              void* d_out, int out_size, void* d_ws, size_t ws_size,
                              hipStream_t stream)
{
    const float* x     = (const float*)d_in[0];   // (1,1024,64,64)
    const float* log_A = (const float*)d_in[1];   // (64,)
    const float* dt_w  = (const float*)d_in[2];   // (1,64)
    const float* dt_b  = (const float*)d_in[3];   // (1,)
    const float* ln_w  = (const float*)d_in[4];   // (64,64)
    const float* ln_b  = (const float*)d_in[5];   // (64,64)
    float* out = (float*)d_out;

    char* ws = (char*)d_ws;
    _Float16* Xh    = (_Float16*)ws;                                   // 8,388,608 B
    _Float16* XhT   = (_Float16*)(ws + (size_t)SEQ * NELEM * 2);       // 8,388,608 B
    float*    log_a = (float*)(ws + (size_t)SEQ * NELEM * 4);          //   262,144 B
    float*    cums  = log_a + (size_t)SEQ * DIM;                       //   262,144 B

    ln_dt_kernel<<<SEQ, 256, 0, stream>>>(x, log_A, dt_w, dt_b, ln_w, ln_b, Xh, log_a);
    scan_kernel<<<DIM, SEQ, 0, stream>>>(log_a, cums);
    transpose_kernel<<<dim3(SEQ / 64, NELEM / 64), 256, 0, stream>>>(Xh, XhT);
    ssm_wmma_kernel<<<dim3(SEQ / 32, DIM), 32, 0, stream>>>(XhT, cums, out);
}
